// HyperMambaSelector_64613488001426
// MI455X (gfx1250) — compile-verified, matrix operand order checked by reading, not run on hardware
//
#include <hip/hip_runtime.h>
#include <math.h>

// ---------------------------------------------------------------------------
// HyperMambaSelector for MI455X (gfx1250, wave32, WMMA + TDM).
// All large GEMMs (in_proj / x_proj / dt_proj / out_proj / selector) run as
// bf16 WMMA (v_wmma_f32_16x16x32_bf16) with fp32 accumulation. GEMM tiles are
// staged into double-buffered LDS by the Tensor Data Mover (tensor_load_to_lds
// with LDS padding = bank-conflict-free 20-dword rows), overlapped with WMMA.
// The sequential selective scan, depthwise conv, LNs and top-k run as VALU.
// ---------------------------------------------------------------------------

#define USE_TDM 1

#define DIMD    1024
#define DSTATE  16
#define DCONV   4
#define DINNER  2048
#define DTRANK  64
#define XDBLW   96
#define XDBLP   128      // padded x_proj output width
#define BATCH   2
#define LOUT    1024
#define LCLS    64
#define MTOK    (BATCH*LOUT)   // 2048 token rows
#define HYPERW  512
#define NSEL    32
#define SELH    256

typedef __bf16 bf16_t;
typedef __attribute__((ext_vector_type(16))) __bf16 v16bf;
typedef __attribute__((ext_vector_type(8)))  float  v8f;
typedef __attribute__((ext_vector_type(4)))  unsigned int v4u;
typedef __attribute__((ext_vector_type(4)))  unsigned int u32x4;
typedef __attribute__((ext_vector_type(8)))  int i32x8;
typedef __attribute__((ext_vector_type(4)))  int i32x4;

union FragU { v4u q[2]; v16bf v; };

__device__ inline bf16_t f2bf(float f) {
    union { float f; unsigned int u; } in; in.f = f;
    unsigned int u = in.u;
    unsigned int r = u + 0x7FFFu + ((u >> 16) & 1u);   // round-to-nearest-even
    union { unsigned short s; bf16_t b; } out;
    out.s = (unsigned short)(r >> 16);
    return out.b;
}
__device__ inline float gelu_f(float x) { return 0.5f * x * (1.f + erff(x * 0.70710678118654752f)); }
__device__ inline float silu_f(float x) { return x / (1.f + __expf(-x)); }

__device__ inline float block_reduce_sum(float v, float* red) {
    int tid = threadIdx.x;
    red[tid] = v; __syncthreads();
    for (int off = 128; off > 0; off >>= 1) {
        if (tid < off) red[tid] += red[tid + off];
        __syncthreads();
    }
    float r = red[0];
    __syncthreads();
    return r;
}

// ---------------------------------------------------------------------------
// TDM: issue one tensor_load_to_lds moving a 128-row x 32-element bf16 tile
// (row stride = ldElems) into LDS at ldsByteOff, with 16B of LDS padding after
// every 64B row so staged rows land on an 80B (20-dword) stride.
// Descriptor packing per CDNA5 ISA 8.3/8.4 (D# group0 128b, group1 256b).
// This toolchain exposes the 6-arg builtin:
//   (u32x4 g0, i32x8 g1, i32x4 g2, i32x4 g3, i32x8 extra, i32 cpol)
// groups 2/3/extra zero-filled for a 2D tensor.
// ---------------------------------------------------------------------------
#if USE_TDM
__device__ inline void tdm_load_tile_128x32(const bf16_t* gptr, int ldElems,
                                            unsigned ldsByteOff)
{
    unsigned long long ga = (unsigned long long)(size_t)gptr;
    u32x4 g0;
    g0[0] = 1u;                                            // count=1, user D#
    g0[1] = ldsByteOff;                                    // lds_addr (bytes)
    g0[2] = (unsigned)(ga & 0xFFFFFFFFu);                  // global_addr[31:0]
    g0[3] = (unsigned)((ga >> 32) & 0x01FFFFFFu)           // global_addr[56:32]
          | (2u << 30);                                    // type=2 ("image")
    i32x8 g1;
    // data_size=1 (2B), pad_enable, pad_interval=3 (16 dwords), pad_amount=3 (4 dwords)
    g1[0] = (1 << 16) | (1 << 20) | (3 << 22) | (3 << 25);
    const unsigned td0 = 32;    // tensor_dim0 (elements per row from tile start)
    const unsigned td1 = 128;   // tensor_dim1 (rows)
    g1[1] = (int)((td0 & 0xFFFFu) << 16);                          // td0[15:0] @63:48
    g1[2] = (int)(((td0 >> 16) & 0xFFFFu) | ((td1 & 0xFFFFu) << 16)); // td0[31:16], td1[15:0]
    g1[3] = (int)(((td1 >> 16) & 0xFFFFu) | (32u << 16));          // td1[31:16], tile_dim0=32
    g1[4] = (int)(128u);                                           // tile_dim1=128, tile_dim2=0
    unsigned long long s0 = (unsigned long long)ldElems;           // tensor_dim0_stride
    g1[5] = (int)(s0 & 0xFFFFFFFFu);
    g1[6] = (int)((s0 >> 32) & 0xFFFFu);                           // stride hi16; dim1_stride=0
    g1[7] = 0;
    i32x4 z4 = {0, 0, 0, 0};                                       // 2D tensor: groups 2/3 zero
    i32x8 z8 = {0, 0, 0, 0, 0, 0, 0, 0};
    __builtin_amdgcn_tensor_load_to_lds(g0, g1, z4, z4, z8, 0);
}
#endif

// ---------------------------------------------------------------------------
// Generic NT GEMM:  C[M,N] = sum_k A[m,k] * W[n,k]   (A: M x K bf16 row-major,
// W: N x K bf16 row-major, C fp32).  M,N multiples of 128, K multiple of 32,
// lda/ldb multiples of 8 (16B-aligned rows).
// Block = 256 threads = 8 waves; tile 128x128, BK=32; wave = 64x32 sub-tile.
// Tiles staged by TDM into double-buffered LDS (20-dword padded rows).
// ---------------------------------------------------------------------------
#define LDST 20   // dwords per staged 32-bf16 row (16 + 4 pad; 80B keeps 16B align)

__global__ __launch_bounds__(256)
void k_gemm_bf16_nt(const bf16_t* __restrict__ A, int lda,
                    const bf16_t* __restrict__ W, int ldb,
                    float* __restrict__ C, int ldc, int K)
{
    __shared__ v4u As4[2][128 * LDST / 4];
    __shared__ v4u Bs4[2][128 * LDST / 4];

    const int tid  = threadIdx.x;
    const int lane = tid & 31;
    const int wave = tid >> 5;
    const int wm   = wave >> 2;   // 0..1 -> 64 rows of M
    const int wn   = wave & 3;    // 0..3 -> 32 cols of N
    const int m0   = blockIdx.y * 128;
    const int n0   = blockIdx.x * 128;

    const bf16_t* Ablk = A + (size_t)m0 * lda;
    const bf16_t* Bblk = W + (size_t)n0 * ldb;

    v8f acc[4][2] = {};
    const int nsteps = K >> 5;

#if USE_TDM
    const unsigned ldsA0 = (unsigned)(size_t)(void*)&As4[0][0];
    const unsigned ldsA1 = (unsigned)(size_t)(void*)&As4[1][0];
    const unsigned ldsB0 = (unsigned)(size_t)(void*)&Bs4[0][0];
    const unsigned ldsB1 = (unsigned)(size_t)(void*)&Bs4[1][0];

    if (wave == 0) {   // prologue: fill buffer 0
        tdm_load_tile_128x32(Ablk, lda, ldsA0);
        tdm_load_tile_128x32(Bblk, ldb, ldsB0);
    }
#else
    const int lrow  = tid >> 1;   // 0..127 staged row
    const int lhalf = tid & 1;    // which 16-bf16 half of the row
#endif

    for (int s = 0; s < nsteps; ++s) {
        const int cur = s & 1;
#if USE_TDM
        if (wave == 0) {
            if (s + 1 < nsteps) {   // kick next K-step into the other buffer
                int k1 = (s + 1) << 5;
                tdm_load_tile_128x32(Ablk + k1, lda, cur ? ldsA0 : ldsA1);
                tdm_load_tile_128x32(Bblk + k1, ldb, cur ? ldsB0 : ldsB1);
                __builtin_amdgcn_s_wait_tensorcnt(2);  // in-order: current pair done
            } else {
                __builtin_amdgcn_s_wait_tensorcnt(0);
            }
        }
        __syncthreads();           // current buffer visible to all waves
#else
        {
            int k0 = s << 5;
            const v4u* ga = (const v4u*)(Ablk + (size_t)lrow * lda + k0);
            v4u a0 = ga[lhalf * 2 + 0];
            v4u a1 = ga[lhalf * 2 + 1];
            const v4u* gb = (const v4u*)(Bblk + (size_t)lrow * ldb + k0);
            v4u b0 = gb[lhalf * 2 + 0];
            v4u b1 = gb[lhalf * 2 + 1];
            unsigned int* As = (unsigned int*)&As4[cur][0];
            unsigned int* Bs = (unsigned int*)&Bs4[cur][0];
            *(v4u*)&As[lrow * LDST + lhalf * 8    ] = a0;
            *(v4u*)&As[lrow * LDST + lhalf * 8 + 4] = a1;
            *(v4u*)&Bs[lrow * LDST + lhalf * 8    ] = b0;
            *(v4u*)&Bs[lrow * LDST + lhalf * 8 + 4] = b1;
        }
        __syncthreads();
#endif

        const unsigned int* As = (const unsigned int*)&As4[cur][0];
        const unsigned int* Bs = (const unsigned int*)&Bs4[cur][0];
        const int fr = lane & 15;
        const int hi = lane >> 4;

        // 16-bit fragment layout: lanes 0-15 carry K {0..7,16..23},
        // lanes 16-31 carry K {8..15,24..31}.
        FragU bfr[2];
        #pragma unroll
        for (int j = 0; j < 2; ++j) {
            int row = wn * 32 + j * 16 + fr;
            bfr[j].q[0] = *(const v4u*)&Bs[row * LDST + hi * 4];
            bfr[j].q[1] = *(const v4u*)&Bs[row * LDST + 8 + hi * 4];
        }
        #pragma unroll
        for (int i = 0; i < 4; ++i) {
            FragU afr;
            int row = wm * 64 + i * 16 + fr;
            afr.q[0] = *(const v4u*)&As[row * LDST + hi * 4];
            afr.q[1] = *(const v4u*)&As[row * LDST + 8 + hi * 4];
            #pragma unroll
            for (int j = 0; j < 2; ++j) {
                acc[i][j] = __builtin_amdgcn_wmma_f32_16x16x32_bf16(
                    false, afr.v, false, bfr[j].v, (short)0, acc[i][j], false, false);
            }
        }
        __syncthreads();           // done reading before buffer is re-filled
    }

    const int fr = lane & 15;
    const int hi = lane >> 4;
    #pragma unroll
    for (int i = 0; i < 4; ++i)
        #pragma unroll
        for (int j = 0; j < 2; ++j)
            #pragma unroll
            for (int r = 0; r < 8; ++r) {
                int row = m0 + wm * 64 + i * 16 + r + hi * 8;
                int col = n0 + wn * 32 + j * 16 + fr;
                C[(size_t)row * ldc + col] = acc[i][j][r];
            }
}

// ---------------------------------------------------------------------------
// Conversions
// ---------------------------------------------------------------------------
__global__ void k_cvt_bf16(const float* __restrict__ in, bf16_t* __restrict__ out, int n)
{
    int i = blockIdx.x * 256 + threadIdx.x;
    if (i < n) out[i] = f2bf(in[i]);
}

// x -> bf16 forward copy + L-flipped copy (per batch) for the backward mamba
__global__ void k_cvt_xflip(const float* __restrict__ x, bf16_t* __restrict__ xf,
                            bf16_t* __restrict__ xb)
{
    int i = blockIdx.x * 256 + threadIdx.x;       // < MTOK*DIMD
    int d = i & (DIMD - 1);
    int m = i >> 10;
    int b = m >> 10, l = m & (LOUT - 1);
    bf16_t v = f2bf(x[i]);
    xf[i] = v;
    xb[((size_t)(b * LOUT + (LOUT - 1 - l))) * DIMD + d] = v;
}

// x_proj (96 x 2048) -> zero-padded (128 x 2048) bf16
__global__ void k_cvt_pad(const float* __restrict__ in, bf16_t* __restrict__ out,
                          int rows, int cols)
{
    int i = blockIdx.x * 256 + threadIdx.x;       // < XDBLP*cols
    int r = i / cols, c = i - r * cols;
    out[i] = (r < rows) ? f2bf(in[(size_t)r * cols + c]) : f2bf(0.f);
}

// ---------------------------------------------------------------------------
// Hypernetwork: pooled = mean(class), h = LN(gelu(W1 h + b1)), hyper = W2 h + b2
// ---------------------------------------------------------------------------
__global__ __launch_bounds__(256)
void k_hyper(const float* __restrict__ cls,
             const float* __restrict__ w1, const float* __restrict__ b1,
             const float* __restrict__ lng, const float* __restrict__ lnb,
             const float* __restrict__ w2, const float* __restrict__ b2,
             float* __restrict__ hyper_out)
{
    __shared__ float pooled[DIMD];
    __shared__ float hbuf[HYPERW];
    __shared__ float red[256];
    int b = blockIdx.x, tid = threadIdx.x;
    for (int d = tid; d < DIMD; d += 256) {
        float s = 0.f;
        for (int j = 0; j < LCLS; ++j) s += cls[((size_t)(b * LCLS + j)) * DIMD + d];
        pooled[d] = s * (1.f / LCLS);
    }
    __syncthreads();
    for (int o = tid; o < HYPERW; o += 256) {
        float a = b1[o];
        const float* wr = w1 + (size_t)o * DIMD;
        for (int d = 0; d < DIMD; ++d) a += wr[d] * pooled[d];
        hbuf[o] = gelu_f(a);
    }
    __syncthreads();
    float s = 0.f;
    for (int o = tid; o < HYPERW; o += 256) s += hbuf[o];
    float mu = block_reduce_sum(s, red) * (1.f / HYPERW);
    float q = 0.f;
    for (int o = tid; o < HYPERW; o += 256) { float t = hbuf[o] - mu; q += t * t; }
    float var = block_reduce_sum(q, red) * (1.f / HYPERW);
    float rs = rsqrtf(var + 1e-5f);
    for (int o = tid; o < HYPERW; o += 256) hbuf[o] = (hbuf[o] - mu) * rs * lng[o] + lnb[o];
    __syncthreads();
    for (int o = tid; o < 2 * DIMD; o += 256) {
        float a = b2[o];
        const float* wr = w2 + (size_t)o * HYPERW;
        for (int j = 0; j < HYPERW; ++j) a += wr[j] * hbuf[j];
        hyper_out[(size_t)b * 2 * DIMD + o] = a;
    }
}

// ---------------------------------------------------------------------------
// Depthwise causal conv (K=4) + bias + silu; emits fp32 and bf16 copies.
// xi = XZ[:, 0:DINNER]
// ---------------------------------------------------------------------------
__global__ void k_conv(const float* __restrict__ XZ, const float* __restrict__ cw,
                       const float* __restrict__ cb, float* __restrict__ xc_f,
                       bf16_t* __restrict__ xc_b)
{
    int idx = blockIdx.x * 256 + threadIdx.x;     // < MTOK*DINNER
    int m = idx >> 11;
    int c = idx & (DINNER - 1);
    int l = m & (LOUT - 1);
    int mb = m - l;                               // batch base row
    float a = cb[c];
    #pragma unroll
    for (int j = 0; j < DCONV; ++j) {
        int li = l - (DCONV - 1) + j;
        if (li >= 0) a += cw[c * DCONV + j] * XZ[(size_t)(mb + li) * (2 * DINNER) + c];
    }
    float v = silu_f(a);
    xc_f[(size_t)m * DINNER + c] = v;
    xc_b[(size_t)m * DINNER + c] = f2bf(v);
}

// softplus(delta_pre + dt_bias), in place
__global__ void k_softplus(float* __restrict__ dlt, const float* __restrict__ bias)
{
    size_t idx = (size_t)blockIdx.x * 256 + threadIdx.x;
    int d = (int)(idx & (DINNER - 1));
    float x = dlt[idx] + bias[d];
    dlt[idx] = (x > 20.f) ? x : log1pf(__expf(x));
}

// ---------------------------------------------------------------------------
// Selective scan: one thread per (batch, channel); 16 states in registers.
// y = (sum_s h_s * C_s + u*D) * silu(z), written as bf16 for out_proj.
// ---------------------------------------------------------------------------
__global__ __launch_bounds__(256)
void k_scan(const float* __restrict__ delta, const float* __restrict__ xdbl,
            const float* __restrict__ xc, const float* __restrict__ XZ,
            const float* __restrict__ A_log, const float* __restrict__ Dp,
            bf16_t* __restrict__ y_out)
{
    int gt = blockIdx.x * 256 + threadIdx.x;      // < BATCH*DINNER
    int b = gt >> 11;
    int d = gt & (DINNER - 1);
    float Arow[DSTATE];
    #pragma unroll
    for (int s = 0; s < DSTATE; ++s) Arow[s] = -__expf(A_log[(size_t)d * DSTATE + s]);
    float h[DSTATE];
    #pragma unroll
    for (int s = 0; s < DSTATE; ++s) h[s] = 0.f;
    float Dd = Dp[d];
    for (int l = 0; l < LOUT; ++l) {
        size_t m = (size_t)b * LOUT + l;
        float dl = delta[m * DINNER + d];
        float u  = xc[m * DINNER + d];
        float du = dl * u;
        const float* bc = xdbl + m * XDBLP + DTRANK;   // [B(16) | C(16)]
        float y = 0.f;
        #pragma unroll
        for (int s = 0; s < DSTATE; ++s) {
            float dA = __expf(dl * Arow[s]);
            h[s] = dA * h[s] + du * bc[s];
            y += h[s] * bc[DSTATE + s];
        }
        y += u * Dd;
        float z = XZ[m * (2 * DINNER) + DINNER + d];
        y *= silu_f(z);
        y_out[m * DINNER + d] = f2bf(y);
    }
}

// mamba_out = out_fwd + flip_L(out_bwd) + 2*(dt_bias + conv_bias)
__global__ void k_combine(const float* __restrict__ of, const float* __restrict__ ob,
                          const float* __restrict__ hyper, float* __restrict__ mo_f,
                          bf16_t* __restrict__ mo_b)
{
    int idx = blockIdx.x * 256 + threadIdx.x;     // < MTOK*DIMD
    int m = idx >> 10;
    int c = idx & (DIMD - 1);
    int b = m >> 10;
    int l = m & (LOUT - 1);
    float v = of[idx]
            + ob[((size_t)(b * LOUT + (LOUT - 1 - l))) * DIMD + c]
            + 2.f * (hyper[(size_t)b * 2 * DIMD + c] + hyper[(size_t)b * 2 * DIMD + DIMD + c]);
    mo_f[idx] = v;
    mo_b[idx] = f2bf(v);
}

// scores[m] = b2 + sum_n w2[n]*gelu(s_pre[m,n] + b1[n])
__global__ __launch_bounds__(256)
void k_score(const float* __restrict__ spre, const float* __restrict__ b1,
             const float* __restrict__ w2, const float* __restrict__ b2,
             float* __restrict__ scores)
{
    __shared__ float red[256];
    int m = blockIdx.x, t = threadIdx.x;
    float v = w2[t] * gelu_f(spre[(size_t)m * SELH + t] + b1[t]);
    float s = block_reduce_sum(v, red);
    if (t == 0) scores[m] = s + b2[0];
}

// ---------------------------------------------------------------------------
// Per-batch: top-32 selection, softmax(scores/0.07), weighted token sum,
// affine LN, 2-layer align MLP -> sel_final[b, 1024]
// ---------------------------------------------------------------------------
__global__ __launch_bounds__(256)
void k_topk_sel(const float* __restrict__ scores, const float* __restrict__ mo,
                const float* __restrict__ lng, const float* __restrict__ lnb,
                const float* __restrict__ aw1, const float* __restrict__ ab1,
                const float* __restrict__ aw2, const float* __restrict__ ab2,
                float* __restrict__ sel_final)
{
    __shared__ float sc[LOUT];
    __shared__ float rv[256]; __shared__ int ri[256];
    __shared__ float topv[NSEL]; __shared__ int topi[NSEL];
    __shared__ float attn[NSEL];
    __shared__ float selv[DIMD];
    __shared__ float h1[DIMD];
    __shared__ float red[256];
    int b = blockIdx.x, tid = threadIdx.x;
    for (int j = tid; j < LOUT; j += 256) sc[j] = scores[b * LOUT + j];
    __syncthreads();
    for (int it = 0; it < NSEL; ++it) {
        float best = -3.4e38f; int bi = 0x7fffffff;
        for (int j = tid; j < LOUT; j += 256) {
            float v = sc[j];
            if (v > best || (v == best && j < bi)) { best = v; bi = j; }
        }
        rv[tid] = best; ri[tid] = bi;
        __syncthreads();
        for (int off = 128; off > 0; off >>= 1) {
            if (tid < off) {
                if (rv[tid + off] > rv[tid] ||
                    (rv[tid + off] == rv[tid] && ri[tid + off] < ri[tid])) {
                    rv[tid] = rv[tid + off]; ri[tid] = ri[tid + off];
                }
            }
            __syncthreads();
        }
        if (tid == 0) { topv[it] = rv[0]; topi[it] = ri[0]; sc[ri[0]] = -3.4e38f; }
        __syncthreads();
    }
    if (tid < NSEL) attn[tid] = __expf((topv[tid] - topv[0]) * (1.f / 0.07f));
    __syncthreads();
    if (tid == 0) {
        float s = 0.f;
        for (int k = 0; k < NSEL; ++k) s += attn[k];
        red[0] = 1.f / s;
    }
    __syncthreads();
    float inv = red[0];
    if (tid < NSEL) attn[tid] *= inv;
    __syncthreads();
    for (int d = tid; d < DIMD; d += 256) {
        float a = 0.f;
        for (int k = 0; k < NSEL; ++k)
            a += attn[k] * mo[((size_t)(b * LOUT + topi[k])) * DIMD + d];
        selv[d] = a;
    }
    __syncthreads();
    float s = 0.f;
    for (int d = tid; d < DIMD; d += 256) s += selv[d];
    float mu = block_reduce_sum(s, red) * (1.f / DIMD);
    float q = 0.f;
    for (int d = tid; d < DIMD; d += 256) { float t = selv[d] - mu; q += t * t; }
    float var = block_reduce_sum(q, red) * (1.f / DIMD);
    float rs = rsqrtf(var + 1e-5f);
    for (int d = tid; d < DIMD; d += 256) selv[d] = (selv[d] - mu) * rs * lng[d] + lnb[d];
    __syncthreads();
    for (int o = tid; o < DIMD; o += 256) {
        float a = ab1[o];
        const float* wr = aw1 + (size_t)o * DIMD;
        for (int d = 0; d < DIMD; ++d) a += wr[d] * selv[d];
        h1[o] = gelu_f(a);
    }
    __syncthreads();
    for (int o = tid; o < DIMD; o += 256) {
        float a = ab2[o];
        const float* wr = aw2 + (size_t)o * DIMD;
        for (int j = 0; j < DIMD; ++j) a += wr[j] * h1[j];
        sel_final[(size_t)b * DIMD + o] = a;
    }
}

// out_row = ln_plain(x_row + gate * addv[batch])  (eps 1e-6, no affine)
__global__ __launch_bounds__(256)
void k_add_ln(const float* __restrict__ x, const float* __restrict__ addv,
              const float* __restrict__ gate, float* __restrict__ out,
              int rowsPerBatch)
{
    __shared__ float red[256];
    int row = blockIdx.x;
    int b = row / rowsPerBatch;
    int tid = threadIdx.x;
    float g = gate[0];
    float v[4];
    float s = 0.f;
    #pragma unroll
    for (int i = 0; i < 4; ++i) {
        int d = tid + i * 256;
        v[i] = x[(size_t)row * DIMD + d] + g * addv[(size_t)b * DIMD + d];
        s += v[i];
    }
    float mu = block_reduce_sum(s, red) * (1.f / DIMD);
    float q = 0.f;
    #pragma unroll
    for (int i = 0; i < 4; ++i) { float t = v[i] - mu; q += t * t; }
    float var = block_reduce_sum(q, red) * (1.f / DIMD);
    float rs = rsqrtf(var + 1e-6f);
    #pragma unroll
    for (int i = 0; i < 4; ++i) {
        int d = tid + i * 256;
        out[(size_t)row * DIMD + d] = (v[i] - mu) * rs;
    }
}

// pooled[b,d] = mean_l aligned_output[b,l,d]
__global__ void k_pool(const float* __restrict__ aligned, float* __restrict__ pooled)
{
    int gt = blockIdx.x * 256 + threadIdx.x;      // < BATCH*DIMD
    int b = gt >> 10, d = gt & (DIMD - 1);
    float s = 0.f;
    for (int l = 0; l < LOUT; ++l) s += aligned[((size_t)(b * LOUT + l)) * DIMD + d];
    pooled[gt] = s * (1.f / LOUT);
}

// ---------------------------------------------------------------------------
// Host driver
// ---------------------------------------------------------------------------
enum {
    IN_X = 0, IN_CLS = 1,
    F_INPROJ = 2, F_CONVW, F_CONVB, F_XPROJ, F_DTW, F_DTB, F_ALOG, F_DSKIP, F_OUTPROJ,
    B_INPROJ = 11, B_CONVW, B_CONVB, B_XPROJ, B_DTW, B_DTB, B_ALOG, B_DSKIP, B_OUTPROJ,
    H_W1 = 20, H_B1, H_LNG, H_LNB, H_W2, H_B2,
    S_W1 = 26, S_B1, S_W2, S_B2,
    AL_LNG = 30, AL_LNB, AL_W1, AL_B1, AL_W2, AL_B2,
    IN_GATEO = 36, IN_GATEC = 37
};

extern "C" void kernel_launch(void* const* d_in, const int* in_sizes, int n_in,
                              void* d_out, int out_size, void* d_ws, size_t ws_size,
                              hipStream_t stream) {
    (void)in_sizes; (void)n_in; (void)out_size; (void)ws_size;
    char* ws = (char*)d_ws;
    size_t off = 0;
    auto alloc = [&](size_t bytes) { size_t o = off; off += (bytes + 255) & ~(size_t)255; return o; };

    float*  hyper   = (float*) (ws + alloc((size_t)BATCH * 2 * DIMD * 4));
    bf16_t* xf      = (bf16_t*)(ws + alloc((size_t)MTOK * DIMD * 2));
    bf16_t* xb      = (bf16_t*)(ws + alloc((size_t)MTOK * DIMD * 2));
    bf16_t* w_in_b  = (bf16_t*)(ws + alloc((size_t)2 * DINNER * DIMD * 2));
    bf16_t* w_xp_b  = (bf16_t*)(ws + alloc((size_t)XDBLP * DINNER * 2));
    bf16_t* w_dt_b  = (bf16_t*)(ws + alloc((size_t)DINNER * DTRANK * 2));
    bf16_t* w_out_b = (bf16_t*)(ws + alloc((size_t)DIMD * DINNER * 2));
    bf16_t* w_s1_b  = (bf16_t*)(ws + alloc((size_t)SELH * DIMD * 2));
    float*  XZ      = (float*) (ws + alloc((size_t)MTOK * 2 * DINNER * 4));
    float*  xc_f    = (float*) (ws + alloc((size_t)MTOK * DINNER * 4));
    bf16_t* xc_b    = (bf16_t*)(ws + alloc((size_t)MTOK * DINNER * 2));
    float*  xdbl_f  = (float*) (ws + alloc((size_t)MTOK * XDBLP * 4));
    bf16_t* xdbl_b  = (bf16_t*)(ws + alloc((size_t)MTOK * XDBLP * 2));
    float*  delta   = (float*) (ws + alloc((size_t)MTOK * DINNER * 4));
    bf16_t* y_b     = (bf16_t*)(ws + alloc((size_t)MTOK * DINNER * 2));
    float*  out_fwd = (float*) (ws + alloc((size_t)MTOK * DIMD * 4));
    float*  out_bwd = (float*) (ws + alloc((size_t)MTOK * DIMD * 4));
    float*  mo_f    = (float*) (ws + alloc((size_t)MTOK * DIMD * 4));
    bf16_t* mo_b    = (bf16_t*)(ws + alloc((size_t)MTOK * DIMD * 2));
    float*  s_pre   = (float*) (ws + alloc((size_t)MTOK * SELH * 4));
    float*  scores  = (float*) (ws + alloc((size_t)MTOK * 4));
    float*  selF    = (float*) (ws + alloc((size_t)BATCH * DIMD * 4));
    float*  pooled  = (float*) (ws + alloc((size_t)BATCH * DIMD * 4));
    // total workspace ~= 140 MB

    const float* x   = (const float*)d_in[IN_X];
    const float* cls = (const float*)d_in[IN_CLS];
    float* out_aligned = (float*)d_out;
    float* out_class   = (float*)d_out + (size_t)MTOK * DIMD;

    // 1) hypernetwork biases
    k_hyper<<<BATCH, 256, 0, stream>>>(cls,
        (const float*)d_in[H_W1], (const float*)d_in[H_B1],
        (const float*)d_in[H_LNG], (const float*)d_in[H_LNB],
        (const float*)d_in[H_W2], (const float*)d_in[H_B2], hyper);

    // 2) bf16 inputs (forward + flipped)
    k_cvt_xflip<<<(MTOK * DIMD) / 256, 256, 0, stream>>>(x, xf, xb);

    // 3) one mamba direction
    auto run_dir = [&](int base, const bf16_t* Xb, float* out_dir) {
        const float* w_in  = (const float*)d_in[base + 0];  // in_proj
        const float* convw = (const float*)d_in[base + 1];
        const float* convb = (const float*)d_in[base + 2];
        const float* xproj = (const float*)d_in[base + 3];
        const float* dtw   = (const float*)d_in[base + 4];
        const float* dtb   = (const float*)d_in[base + 5];
        const float* alog  = (const float*)d_in[base + 6];
        const float* dsk   = (const float*)d_in[base + 7];
        const float* outw  = (const float*)d_in[base + 8];

        k_cvt_bf16<<<(2 * DINNER * DIMD) / 256, 256, 0, stream>>>(w_in, w_in_b, 2 * DINNER * DIMD);
        k_cvt_pad <<<(XDBLP * DINNER) / 256, 256, 0, stream>>>(xproj, w_xp_b, XDBLW, DINNER);
        k_cvt_bf16<<<(DINNER * DTRANK) / 256, 256, 0, stream>>>(dtw, w_dt_b, DINNER * DTRANK);
        k_cvt_bf16<<<(DIMD * DINNER) / 256, 256, 0, stream>>>(outw, w_out_b, DIMD * DINNER);

        // in_proj: (2048 x 1024) @ (4096 x 1024)^T
        k_gemm_bf16_nt<<<dim3(32, 16), 256, 0, stream>>>(Xb, DIMD, w_in_b, DIMD, XZ, 2 * DINNER, DIMD);
        // depthwise conv + silu
        k_conv<<<(MTOK * DINNER) / 256, 256, 0, stream>>>(XZ, convw, convb, xc_f, xc_b);
        // x_proj: (2048 x 2048) @ (128 x 2048)^T (padded)
        k_gemm_bf16_nt<<<dim3(1, 16), 256, 0, stream>>>(xc_b, DINNER, w_xp_b, DINNER, xdbl_f, XDBLP, DINNER);
        k_cvt_bf16<<<(MTOK * XDBLP) / 256, 256, 0, stream>>>(xdbl_f, xdbl_b, MTOK * XDBLP);
        // dt_proj: (2048 x 64) @ (2048 x 64)^T
        k_gemm_bf16_nt<<<dim3(16, 16), 256, 0, stream>>>(xdbl_b, XDBLP, w_dt_b, DTRANK, delta, DINNER, DTRANK);
        k_softplus<<<(MTOK * DINNER) / 256, 256, 0, stream>>>(delta, dtb);
        // sequential selective scan
        k_scan<<<(BATCH * DINNER) / 256, 256, 0, stream>>>(delta, xdbl_f, xc_f, XZ, alog, dsk, y_b);
        // out_proj: (2048 x 2048) @ (1024 x 2048)^T
        k_gemm_bf16_nt<<<dim3(8, 16), 256, 0, stream>>>(y_b, DINNER, w_out_b, DINNER, out_dir, DIMD, DINNER);
    };

    run_dir(F_INPROJ, xf, out_fwd);
    run_dir(B_INPROJ, xb, out_bwd);

    // 4) combine fwd + flipped bwd + hyper biases
    k_combine<<<(MTOK * DIMD) / 256, 256, 0, stream>>>(out_fwd, out_bwd, hyper, mo_f, mo_b);

    // 5) selector: (2048 x 1024) @ (256 x 1024)^T, then gelu+dot -> scores
    k_cvt_bf16<<<(SELH * DIMD) / 256, 256, 0, stream>>>((const float*)d_in[S_W1], w_s1_b, SELH * DIMD);
    k_gemm_bf16_nt<<<dim3(2, 16), 256, 0, stream>>>(mo_b, DIMD, w_s1_b, DIMD, s_pre, SELH, DIMD);
    k_score<<<MTOK, 256, 0, stream>>>(s_pre, (const float*)d_in[S_B1],
                                      (const float*)d_in[S_W2], (const float*)d_in[S_B2], scores);

    // 6) top-32 + softmax + weighted sum + LN + align MLP
    k_topk_sel<<<BATCH, 256, 0, stream>>>(scores, mo_f,
        (const float*)d_in[AL_LNG], (const float*)d_in[AL_LNB],
        (const float*)d_in[AL_W1], (const float*)d_in[AL_B1],
        (const float*)d_in[AL_W2], (const float*)d_in[AL_B2], selF);

    // 7) aligned_output, pooled, aligned_class
    k_add_ln<<<MTOK, 256, 0, stream>>>(x, selF, (const float*)d_in[IN_GATEO], out_aligned, LOUT);
    k_pool<<<(BATCH * DIMD) / 256, 256, 0, stream>>>(out_aligned, pooled);
    k_add_ln<<<BATCH * LCLS, 256, 0, stream>>>(cls, pooled, (const float*)d_in[IN_GATEC], out_class, LCLS);
}
